// OCRModel_50714973831896
// MI455X (gfx1250) — compile-verified, matrix operand order checked by reading, not run on hardware
//
#include <hip/hip_runtime.h>
#include <hip/hip_bf16.h>
#include <math.h>

// ---------------------------------------------------------------------------
// Types for CDNA5 WMMA (wave32): v_wmma_f32_16x16x32_bf16
// ---------------------------------------------------------------------------
typedef __attribute__((ext_vector_type(16))) __bf16 v16bf;
typedef __attribute__((ext_vector_type(8)))  __bf16 v8bf;
typedef __attribute__((ext_vector_type(8)))  float  v8f;

#define B_    16
#define E_    1024
#define H_    2048
#define V_    255
#define VP_   256           // padded V

#define NBLK  36
#define TPB   128           // 4 waves/block -> 144 waves total

__device__ __forceinline__ v8f wmma_bf16(v16bf a, v16bf b, v8f c) {
  // D = A(16x32 bf16) * B(32x16 bf16) + C(16x16 f32)
  return __builtin_amdgcn_wmma_f32_16x16x32_bf16(
      /*neg_a=*/false, a, /*neg_b=*/false, b,
      /*c_mod=*/(short)0, c, /*reuse_a=*/false, /*reuse_b=*/false);
}

// A-fragment (16x32 bf16, row-major source, lane-layout per ISA 7.12.2):
// lane L: m = L&15, K-halves at +0/+16, 8 contiguous bf16 each -> two b128 loads.
__device__ __forceinline__ v16bf load_a_frag(const __bf16* __restrict__ p) {
  v8bf lo = *(const v8bf*)(p);
  v8bf hi = *(const v8bf*)(p + 16);
  v16bf a;
#pragma unroll
  for (int i = 0; i < 8; ++i) { a[i] = lo[i]; a[i + 8] = hi[i]; }
  return a;
}

// B-fragments are pre-packed: fragment f stores 32 lanes x 16 bf16 contiguous.
__device__ __forceinline__ v16bf load_b_frag(const __bf16* __restrict__ pack,
                                             int f, int lane) {
  return *(const v16bf*)(pack + ((size_t)f << 9) + (lane << 4));
}

// ---------------------------------------------------------------------------
// Weight pack: fp32 row-major [rows x cols] (ld = row stride) -> bf16 WMMA
// B-fragment order. k-map per ISA: k = 16*(e/8) + (e%8) + 8*(lane/16).
// Zero-pads out-of-range rows/cols (V=255 -> 256).
// ---------------------------------------------------------------------------
__global__ void pack_weights(const float* __restrict__ src, __bf16* __restrict__ dst,
                             int KT, int NT, int rows, int cols, int ld) {
  int total = KT * NT * 512;
  for (int o = blockIdx.x * blockDim.x + threadIdx.x; o < total;
       o += gridDim.x * blockDim.x) {
    int f = o >> 9, r = o & 511;
    int lane = r >> 4, e = r & 15;
    int ktile = f / NT, ntile = f - ktile * NT;
    int k = ktile * 32 + ((e >> 3) << 4) + (e & 7) + ((lane >> 4) << 3);
    int n = ntile * 16 + (lane & 15);
    float v = (k < rows && n < cols) ? src[(size_t)k * ld + n] : 0.0f;
    dst[o] = (__bf16)v;
  }
}

// embeddings fp32 -> bf16; also (re)initialize the grid barrier each call.
__global__ void cvt_emb_init(const float* __restrict__ src, __bf16* __restrict__ dst,
                             int n, unsigned* __restrict__ bar) {
  int i = blockIdx.x * blockDim.x + threadIdx.x;
  if (i < n) dst[i] = (__bf16)src[i];
  if (i == 0) { bar[0] = 0u; bar[1] = 0u; }
}

// ---------------------------------------------------------------------------
// Device-wide sense barrier (persistent grid, all NBLK blocks co-resident).
// ---------------------------------------------------------------------------
__device__ __forceinline__ void grid_sync(unsigned* __restrict__ bar) {
  __syncthreads();
  if (threadIdx.x == 0) {
    __threadfence();
    unsigned gen = __atomic_load_n(bar + 1, __ATOMIC_RELAXED);
    if (atomicAdd(bar, 1u) == (unsigned)(NBLK - 1)) {
      __atomic_store_n(bar, 0u, __ATOMIC_RELAXED);
      __atomic_fetch_add(bar + 1, 1u, __ATOMIC_RELEASE);
    } else {
      while (__atomic_load_n(bar + 1, __ATOMIC_ACQUIRE) == gen)
        __builtin_amdgcn_s_sleep(2);
    }
  }
  __syncthreads();
}

// ---------------------------------------------------------------------------
// Persistent RNN decode. 144 waves:
//   waves   0..127 : one Wll column tile each (h @ Wll, then + out @ Wol)
//   waves 128..143 : one Wlo column tile each (logits path)
// State h kept in bf16 (fp32 accumulate inside WMMA), weights stream from L2.
// ---------------------------------------------------------------------------
__global__ void __launch_bounds__(TPB, 1)
rnn_persistent(const __bf16* __restrict__ Wt_p,  const __bf16* __restrict__ Wll_p,
               const __bf16* __restrict__ Wlo_p, const __bf16* __restrict__ Wol_p,
               const __bf16* __restrict__ emb_b, __bf16* __restrict__ h_b,
               __bf16* __restrict__ out_b,
               const float* __restrict__ bt,  const float* __restrict__ bll,
               const float* __restrict__ blo, const float* __restrict__ bol,
               float* __restrict__ dout, int T, unsigned* __restrict__ bar) {
  const int wave  = (blockIdx.x * TPB + threadIdx.x) >> 5;
  const int lane  = threadIdx.x & 31;
  const int m     = lane & 15;          // C/D column index == output col within tile
  const int khalf = (lane >> 4) << 3;   // A-fragment K-half select
  const int rbase = (lane >> 4) << 3;   // C/D rows: rbase + i

  // ---- h0 = emb @ Wt + bt (linear) -> h_b (bf16) --------------------------
  if (wave < 128) {
    const int nt = wave;
    v8f acc = {0.f, 0.f, 0.f, 0.f, 0.f, 0.f, 0.f, 0.f};
    const __bf16* arow = emb_b + (size_t)m * E_ + khalf;
#pragma unroll 4
    for (int kt = 0; kt < E_ / 32; ++kt) {
      v16bf a = load_a_frag(arow + kt * 32);
      v16bf b = load_b_frag(Wt_p, kt * 128 + nt, lane);
      acc = wmma_bf16(a, b, acc);
    }
    const int n = nt * 16 + m;
    const float bb = bt[n];
#pragma unroll
    for (int i = 0; i < 8; ++i)
      h_b[(size_t)(rbase + i) * H_ + n] = (__bf16)(acc[i] + bb);
  }
  grid_sync(bar);

  // ---- decode loop --------------------------------------------------------
  for (int t = 0; t < T; ++t) {
    v8f acc = {0.f, 0.f, 0.f, 0.f, 0.f, 0.f, 0.f, 0.f};
    const __bf16* hrow = h_b + (size_t)m * H_ + khalf;

    if (wave < 128) {
      // acc = (h @ Wll) tile  (64 K-steps of 32)
      const int nt = wave;
#pragma unroll 4
      for (int kt = 0; kt < H_ / 32; ++kt) {
        const __bf16* bp = Wll_p + (((size_t)(kt * 128 + nt)) << 9) + (lane << 4);
        __builtin_prefetch(bp + (128 << 9), 0, 1);   // next K-panel -> L2/L0
        v16bf a = load_a_frag(hrow + kt * 32);
        v16bf b = *(const v16bf*)bp;
        acc = wmma_bf16(a, b, acc);
      }
    } else {
      // logits tile: out = tanh(h @ Wlo + blo); write logits + bf16 copy
      const int nt = wave - 128;
#pragma unroll 4
      for (int kt = 0; kt < H_ / 32; ++kt) {
        v16bf a = load_a_frag(hrow + kt * 32);
        v16bf b = load_b_frag(Wlo_p, kt * 16 + nt, lane);
        acc = wmma_bf16(a, b, acc);
      }
      const int n = nt * 16 + m;
      const float bb = (n < V_) ? blo[n] : 0.0f;
#pragma unroll
      for (int i = 0; i < 8; ++i) {
        float v = tanhf(acc[i] + bb);
        if (n < V_)
          dout[((size_t)(rbase + i) * T + t) * V_ + n] = v;  // [B,T,V] logits
        out_b[(rbase + i) * VP_ + n] = (__bf16)v;
      }
    }
    grid_sync(bar);   // out_b ready

    if (wave < 128) {
      // acc += (out @ Wol) tile (8 K-steps over padded V=256), bias, tanh -> h
      const int nt = wave;
      const __bf16* orow = out_b + (size_t)m * VP_ + khalf;
#pragma unroll
      for (int kt = 0; kt < VP_ / 32; ++kt) {
        v16bf a = load_a_frag(orow + kt * 32);
        v16bf b = load_b_frag(Wol_p, kt * 128 + nt, lane);
        acc = wmma_bf16(a, b, acc);
      }
      const int n = nt * 16 + m;
      const float bb = bll[n] + bol[n];
#pragma unroll
      for (int i = 0; i < 8; ++i) {
        float v = tanhf(acc[i] + bb);
        h_b[(size_t)(rbase + i) * H_ + n] = (__bf16)v;
      }
    }
    grid_sync(bar);   // h_b ready for next step
  }
}

// ---------------------------------------------------------------------------
// Softmax over V=255, one wave per (b,t) row, in place on d_out.
// ---------------------------------------------------------------------------
__global__ void softmax_rows(float* __restrict__ x, int rows) {
  const int wid  = (blockIdx.x * blockDim.x + threadIdx.x) >> 5;
  const int lane = threadIdx.x & 31;
  if (wid >= rows) return;
  float* row = x + (size_t)wid * V_;
  float vals[8];
  float mx = -3.0e38f;
#pragma unroll
  for (int j = 0; j < 8; ++j) {
    int v = lane + 32 * j;
    vals[j] = (v < V_) ? row[v] : -3.0e38f;
    mx = fmaxf(mx, vals[j]);
  }
#pragma unroll
  for (int off = 16; off > 0; off >>= 1) mx = fmaxf(mx, __shfl_xor(mx, off, 32));
  float s = 0.0f;
#pragma unroll
  for (int j = 0; j < 8; ++j) {
    int v = lane + 32 * j;
    vals[j] = (v < V_) ? __expf(vals[j] - mx) : 0.0f;
    s += vals[j];
  }
#pragma unroll
  for (int off = 16; off > 0; off >>= 1) s += __shfl_xor(s, off, 32);
  const float inv = 1.0f / s;
#pragma unroll
  for (int j = 0; j < 8; ++j) {
    int v = lane + 32 * j;
    if (v < V_) row[v] = vals[j] * inv;
  }
}

// ---------------------------------------------------------------------------
// Host-side launcher (graph-capture safe: only async launches on `stream`).
// Input order: embeddings, Wt, bt, Wll, bll, Wlo, blo, Wol, bol, cap_length
// ---------------------------------------------------------------------------
extern "C" void kernel_launch(void* const* d_in, const int* in_sizes, int n_in,
                              void* d_out, int out_size, void* d_ws, size_t ws_size,
                              hipStream_t stream) {
  (void)n_in; (void)ws_size; (void)in_sizes;
  const float* emb = (const float*)d_in[0];
  const float* Wt  = (const float*)d_in[1];
  const float* bt  = (const float*)d_in[2];
  const float* Wll = (const float*)d_in[3];
  const float* bll = (const float*)d_in[4];
  const float* Wlo = (const float*)d_in[5];
  const float* blo = (const float*)d_in[6];
  const float* Wol = (const float*)d_in[7];
  const float* bol = (const float*)d_in[8];
  float* dout = (float*)d_out;

  const int T = out_size / (B_ * V_);   // cap_length, derived deterministically

  // Workspace carve-up (bf16 packed weights + state + barrier), ~14.2 MiB.
  char* w = (char*)d_ws;
  __bf16* Wt_p  = (__bf16*)w;  w += (size_t)32 * 128 * 512 * 2;  // 4 MiB
  __bf16* Wll_p = (__bf16*)w;  w += (size_t)64 * 128 * 512 * 2;  // 8 MiB
  __bf16* Wlo_p = (__bf16*)w;  w += (size_t)64 * 16  * 512 * 2;  // 1 MiB
  __bf16* Wol_p = (__bf16*)w;  w += (size_t)8  * 128 * 512 * 2;  // 1 MiB
  __bf16* emb_b = (__bf16*)w;  w += (size_t)B_ * E_ * 2;
  __bf16* h_b   = (__bf16*)w;  w += (size_t)B_ * H_ * 2;
  __bf16* out_b = (__bf16*)w;  w += (size_t)B_ * VP_ * 2;
  unsigned* bar = (unsigned*)w;

  // 1) pack weights to bf16 WMMA fragment order (one-time per call; cheap).
  {
    int tot, blk = 256;
    tot = 32 * 128 * 512;
    pack_weights<<<(tot + blk - 1) / blk, blk, 0, stream>>>(Wt,  Wt_p,  32, 128, E_, H_, H_);
    tot = 64 * 128 * 512;
    pack_weights<<<(tot + blk - 1) / blk, blk, 0, stream>>>(Wll, Wll_p, 64, 128, H_, H_, H_);
    tot = 64 * 16 * 512;
    pack_weights<<<(tot + blk - 1) / blk, blk, 0, stream>>>(Wlo, Wlo_p, 64, 16,  H_, V_, V_);
    tot = 8 * 128 * 512;
    pack_weights<<<(tot + blk - 1) / blk, blk, 0, stream>>>(Wol, Wol_p, 8,  128, V_, H_, H_);
    cvt_emb_init<<<(B_ * E_ + blk - 1) / blk, blk, 0, stream>>>(emb, emb_b, B_ * E_, bar);
  }

  // 2) persistent recurrent decode (all 512 steps inside one kernel).
  rnn_persistent<<<NBLK, TPB, 0, stream>>>(Wt_p, Wll_p, Wlo_p, Wol_p,
                                           emb_b, h_b, out_b,
                                           bt, bll, blo, bol, dout, T, bar);

  // 3) softmax epilogue over V, one wave per (b,t) row.
  {
    int rows = B_ * T;
    int waves_per_blk = 256 / 32;
    int blocks = (rows + waves_per_blk - 1) / waves_per_blk;
    softmax_rows<<<blocks, 256, 0, stream>>>(dout, rows);
  }
}
